// LinearAttn_36601711296952
// MI455X (gfx1250) — matrix operand order repaired and checked by reading, not compiled, not run on hardware
//
#include <hip/hip_runtime.h>

#define B_   2
#define S_   8192
#define H_   16
#define QK_SCALE_ 0.08838834764831845f   /* 1/sqrt(128) */
#define EPS_ 1e-5f

typedef unsigned int       u32;
typedef unsigned long long u64;
typedef __attribute__((ext_vector_type(16))) __bf16 v16bf;
typedef __attribute__((ext_vector_type(8)))  float  v8f;
typedef __attribute__((ext_vector_type(4)))  u32    v4u;
typedef __attribute__((ext_vector_type(8)))  int    v8i;
typedef __attribute__((ext_vector_type(4)))  int    v4i;

#define WMMA_BF16(acc, a, b)                                                     \
  (acc) = __builtin_amdgcn_wmma_f32_16x16x32_bf16(false, (a), false, (b),        \
                                                  (short)0, (acc), false, false)

#define ZERO8x8(acc)                                                             \
  _Pragma("unroll") for (int _n = 0; _n < 8; ++_n)                               \
      _Pragma("unroll") for (int _r = 0; _r < 8; ++_r)(acc)[_n][_r] = 0.f

// ---------------- Tensor Data Mover: 2-D bf16 tile -> LDS (ISA ch.8 D#) ----------------
// Flat shared-aperture pointers carry the LDS byte offset in the low 32 bits (ISA 10.2).
__device__ __forceinline__ u32 lds_addr_of(const void* p) { return (u32)(u64)p; }

__device__ __forceinline__ void tdm_load_2d(u32 lds_byte, u64 gaddr, u32 stride_elems,
                                            u32 tile_d0, u32 tile_d1,
                                            u32 pad_int_code, u32 pad_amt_code,
                                            u32 tensor_d0, u32 tensor_d1) {
  v4u g0;
  g0[0] = 1u;                                  // count=1 (valid), user mode
  g0[1] = lds_byte;                            // lds_addr
  g0[2] = (u32)gaddr;                          // global_addr[31:0]
  g0[3] = (u32)(gaddr >> 32) | (2u << 30);     // global_addr[56:32] | type=2 ("image")
  v8i g1;
  g1[0] = (int)((1u << 16) |                   // data_size = 2 bytes
                (1u << 20) |                   // pad_enable
                (pad_int_code << 22) | (pad_amt_code << 25));
  g1[1] = (int)((tensor_d0 & 0xFFFFu) << 16);                  // tensor_dim0 lo
  g1[2] = (int)((tensor_d0 >> 16) | ((tensor_d1 & 0xFFFFu) << 16));
  g1[3] = (int)((tensor_d1 >> 16) | (tile_d0 << 16));          // tile_dim0
  g1[4] = (int)tile_d1;                                        // tile_dim1 (tile_dim2=0)
  g1[5] = (int)stride_elems;                                   // tensor_dim0_stride lo
  g1[6] = 0;
  g1[7] = 0;
  const v4i z4 = {0, 0, 0, 0};
#if __clang_major__ >= 23
  const v8i z8 = {0, 0, 0, 0, 0, 0, 0, 0};
  __builtin_amdgcn_tensor_load_to_lds(g0, g1, z4, z4, z8, 0);
#else
  __builtin_amdgcn_tensor_load_to_lds(g0, g1, z4, z4, 0);
#endif
}

// ---------------- WMMA fragment loaders (CDNA5 ISA 7.12.2, wave32) ----------------
// A fragment 16x32 from row-major [M][K]: contiguous pairs -> ds_load_b128.
__device__ __forceinline__ v16bf frag_a_rm(const __bf16* p, int stride, int rowbase,
                                           int kbase, int lane) {
  v16bf f;
  const __bf16* rp = p + (long)(rowbase + (lane & 15)) * stride + kbase;
  const int h8 = (lane >> 4) << 4;
#pragma unroll
  for (int e = 0; e < 16; ++e) {
    const int r = e >> 1, s = e & 1;
    f[e] = rp[h8 + (r << 1) + s];
  }
  return f;
}

// B fragment 32x16 from [N][K] row-major storage: contiguous -> ds_load_b128.
__device__ __forceinline__ v16bf frag_b_nk(const __bf16* p, int stride, int kbase,
                                           int nbase, int lane) {
  v16bf f;
  const __bf16* rp = p + (long)(nbase + (lane & 15)) * stride + kbase + ((lane >> 4) << 4);
#pragma unroll
  for (int e = 0; e < 16; ++e) f[e] = rp[e];
  return f;
}

// Scale a bf16 fragment by a scalar using packed bf16 math (v_pk_mul_bf16).
__device__ __forceinline__ v16bf scale_bf16(v16bf v, float s) {
  const __bf16 sb = (__bf16)s;
  v16bf sp;
#pragma unroll
  for (int e = 0; e < 16; ++e) sp[e] = sb;
  return v * sp;
}

// ---------------- f32 -> bf16 bulk convert (4 elems/thread) ----------------
__global__ __launch_bounds__(256) void f32_to_bf16_kernel(const float* __restrict__ src,
                                                          __bf16* __restrict__ dst) {
  const long i = (((long)blockIdx.x * 256) + threadIdx.x) << 2;
  const float4 v = *(const float4*)(src + i);
  __bf16 t[4] = {(__bf16)v.x, (__bf16)v.y, (__bf16)v.z, (__bf16)v.w};
  *(uint2*)(dst + i) = *(const uint2*)t;
}

// ---------------- TDM double-buffered WMMA GEMM: C = A[M,K] * W[N,K]^T (+bias) --------
template <typename TC>
__global__ __launch_bounds__(256) void wmma_gemm(const __bf16* __restrict__ A, int lda,
                                                 const __bf16* __restrict__ W,
                                                 const float* __restrict__ bias,
                                                 TC* __restrict__ C, int ldc, int K) {
  __shared__ __bf16 As[2][128][40];   // 64B rows + 16B TDM pad = 80B stride
  __shared__ __bf16 Bs[2][128][40];
  const int tid = threadIdx.x;
  const int lane = tid & 31, wave = tid >> 5;
  const long m0 = (long)blockIdx.y * 128;
  const long n0 = (long)blockIdx.x * 128;
  const int nk = K >> 5;

  v8f acc[8];
  ZERO8x8(acc);

  if (wave == 0) {
    tdm_load_2d(lds_addr_of(&As[0][0][0]), (u64)(A + m0 * lda), (u32)lda,
                32, 128, /*pad: every 16 DW add 4 DW*/ 3, 3, (u32)lda, 16384u);
    tdm_load_2d(lds_addr_of(&Bs[0][0][0]), (u64)(W + n0 * (long)K), (u32)K,
                32, 128, 3, 3, (u32)K, 16384u);
  }
  for (int kt = 0; kt < nk; ++kt) {
    const int buf = kt & 1;
    if (wave == 0) __builtin_amdgcn_s_wait_tensorcnt(0);  // tiles for `buf` landed
    __syncthreads();                                      // publish; prev buf free
    if (wave == 0 && kt + 1 < nk) {
      const long k1 = (long)(kt + 1) << 5;
      tdm_load_2d(lds_addr_of(&As[buf ^ 1][0][0]), (u64)(A + m0 * lda + k1), (u32)lda,
                  32, 128, 3, 3, (u32)lda, 16384u);
      tdm_load_2d(lds_addr_of(&Bs[buf ^ 1][0][0]), (u64)(W + n0 * (long)K + k1), (u32)K,
                  32, 128, 3, 3, (u32)K, 16384u);
    }
    const v16bf a = frag_a_rm(&As[buf][0][0], 40, wave << 4, 0, lane);
#pragma unroll
    for (int n = 0; n < 8; ++n) {
      const v16bf b = frag_b_nk(&Bs[buf][0][0], 40, 0, n << 4, lane);
      WMMA_BF16(acc[n], a, b);
    }
  }

  const long row = m0 + (wave << 4) + ((lane >> 4) << 3);
  const int coll = lane & 15;
#pragma unroll
  for (int n = 0; n < 8; ++n) {
    const long col = n0 + (n << 4) + coll;
    const float bv = bias ? bias[col] : 0.f;
#pragma unroll
    for (int r = 0; r < 8; ++r)
      C[(row + r) * (long)ldc + col] = (TC)(acc[n][r] + bv);
  }
}

// ---------------- Lightning attention (transposed tiles): block per (b,h) ----------------
// All products are computed transposed so every WMMA operand is a contiguous
// ds_load_b128 fragment; the k*kdec operand is transposed+scaled into the (momentarily
// dead) state-mirror LDS region between the two mask-window barriers.
__global__ __launch_bounds__(256) void lightning_attn(const __bf16* __restrict__ Q,
                                                      const __bf16* __restrict__ Kx,
                                                      const __bf16* __restrict__ V,
                                                      float* __restrict__ O) {
  __shared__ __bf16 qs[128][136];    // q tile [i][d] (TDM); later masked scores ss[i][j]
  __shared__ __bf16 ks[128][136];    // raw k tile [j][d] (TDM)
  __shared__ __bf16 vsT[128][136];   // v^T tile [e][j] (manual transposed stage)
  __shared__ __bf16 SbT[128][136];   // state^T mirror [e][d]; mid-chunk: ksT[d][j]*kdec
  __shared__ float  kdec[128];       // QK_SCALE * exp(-slope*(127-j))
  __shared__ float  qdtab[128];      // exp(-slope*(i+1))
  __shared__ float  mtab[128];       // QK_SCALE * exp(-slope*diff)

  const int tid = threadIdx.x;
  const int lane = tid & 31, wave = tid >> 5;
  const int bh = blockIdx.x;
  const int b = bh >> 4, h = bh & 15;
  const float slope = -logf(1.0f - exp2f(-5.0f - (float)h));
  const float lam = expf(-slope * 128.0f);
  const long hcol = (long)h * 128;

  if (tid < 128) {
    kdec[tid] = QK_SCALE_ * expf(-slope * (float)(127 - tid));
    qdtab[tid] = expf(-slope * ((float)tid + 1.0f));
    mtab[tid] = QK_SCALE_ * expf(-slope * (float)tid);
  }
  for (int idx = tid; idx < 128 * 136; idx += 256) (&SbT[0][0])[idx] = (__bf16)0.f;

  v8f Sreg[8];                       // f32 state^T strip, rows e in [wave*16, +16)
  ZERO8x8(Sreg);

  const int e0 = wave << 4;          // this wave's e-strip (and j-strip for scores)
  const int m15 = lane & 15;
  const int rhi = (lane >> 4) << 3;

  __syncthreads();

  for (int c = 0; c < 64; ++c) {
    const long rowg0 = (long)b * S_ + (long)c * 128;

    // ---- TDM q/k tiles; overlap with the manual transposed stage of v ----
    if (wave == 0) {
      tdm_load_2d(lds_addr_of(&qs[0][0]), (u64)(Q + rowg0 * 2048 + hcol), 2048u,
                  128, 128, 5, 3, 2048u, 16384u);
      tdm_load_2d(lds_addr_of(&ks[0][0]), (u64)(Kx + rowg0 * 2048 + hcol), 2048u,
                  128, 128, 5, 3, 2048u, 16384u);
    }
    {
      const int eg = (tid & 31) << 2;            // 4 consecutive e per thread
      const int jb = tid >> 5;
#pragma unroll
      for (int p = 0; p < 16; ++p) {
        const int j = jb + (p << 3);
        __bf16 t[4];
        *(uint2*)t = *(const uint2*)(V + (rowg0 + j) * 6144 + hcol + eg);
#pragma unroll
        for (int u = 0; u < 4; ++u) vsT[eg + u][j] = t[u];
      }
    }
    if (wave == 0) __builtin_amdgcn_s_wait_tensorcnt(0);
    __syncthreads();

    // ---- scores^T[j][i] = k @ q^T ----
    v8f accS[8];
    ZERO8x8(accS);
#pragma unroll
    for (int kt = 0; kt < 4; ++kt) {
      const v16bf a = frag_a_rm(&ks[0][0], 136, e0, kt << 5, lane);
#pragma unroll
      for (int n = 0; n < 8; ++n) {
        const v16bf bb = frag_b_nk(&qs[0][0], 136, kt << 5, n << 4, lane);
        WMMA_BF16(accS[n], a, bb);
      }
    }

    // ---- inter^T[e][i] = S^T @ (q * qdec)^T   (reads qs/SbT before mask window) ----
    float qdv[8];
#pragma unroll
    for (int n = 0; n < 8; ++n) qdv[n] = qdtab[(n << 4) + m15];
    v8f accO[8];
    ZERO8x8(accO);
#pragma unroll
    for (int kt = 0; kt < 4; ++kt) {
      const v16bf a = frag_a_rm(&SbT[0][0], 136, e0, kt << 5, lane);
#pragma unroll
      for (int n = 0; n < 8; ++n) {
        const v16bf bb = scale_bf16(frag_b_nk(&qs[0][0], 136, kt << 5, n << 4, lane),
                                    qdv[n]);
        WMMA_BF16(accO[n], a, bb);
      }
    }

    // ---- mask window: park ss[i][j] in q tile; build ksT*kdec in SbT region ----
    __syncthreads();
#pragma unroll
    for (int n = 0; n < 8; ++n) {
#pragma unroll
      for (int r = 0; r < 8; ++r) {
        const int j = e0 + rhi + r;
        const int i = (n << 4) + m15;
        const float m = (i >= j) ? mtab[i - j] : 0.f;
        qs[i][j] = (__bf16)(accS[n][r] * m);
      }
    }
    {
      const int dg = (tid & 31) << 2;            // 4 consecutive d per thread
      const int jb = tid >> 5;
#pragma unroll
      for (int p = 0; p < 16; ++p) {
        const int j = jb + (p << 3);
        const float kd = kdec[j];
        __bf16 t[4];
        *(uint2*)t = *(const uint2*)&ks[j][dg];
#pragma unroll
        for (int u = 0; u < 4; ++u) SbT[dg + u][j] = (__bf16)((float)t[u] * kd);
      }
    }
    __syncthreads();

    // ---- intra: out^T[e][i] += v^T @ ss ; store chunk output (transposed scatter) ----
#pragma unroll
    for (int kt = 0; kt < 4; ++kt) {
      const v16bf a = frag_a_rm(&vsT[0][0], 136, e0, kt << 5, lane);
#pragma unroll
      for (int n = 0; n < 8; ++n) {
        const v16bf bb = frag_b_nk(&qs[0][0], 136, kt << 5, n << 4, lane);
        WMMA_BF16(accO[n], a, bb);
      }
    }
#pragma unroll
    for (int n = 0; n < 8; ++n)
#pragma unroll
      for (int r = 0; r < 8; ++r)
        O[(rowg0 + (n << 4) + m15) * 2048 + hcol + e0 + rhi + r] = accO[n][r];

    // ---- state update: S^T = lam*S^T + v^T @ (k * kdec)   (f32 in registers) ----
#pragma unroll
    for (int n = 0; n < 8; ++n)
#pragma unroll
      for (int r = 0; r < 8; ++r) Sreg[n][r] *= lam;
#pragma unroll
    for (int kt = 0; kt < 4; ++kt) {
      const v16bf a = frag_a_rm(&vsT[0][0], 136, e0, kt << 5, lane);
#pragma unroll
      for (int n = 0; n < 8; ++n) {
        const v16bf bb = frag_b_nk(&SbT[0][0], 136, kt << 5, n << 4, lane);  // ksT*kdec
        WMMA_BF16(Sreg[n], a, bb);
      }
    }
    __syncthreads();   // all waves done reading SbT(ksT)/qs/vsT for this chunk
#pragma unroll
    for (int n = 0; n < 8; ++n)
#pragma unroll
      for (int r = 0; r < 8; ++r)
        SbT[e0 + rhi + r][(n << 4) + m15] = (__bf16)Sreg[n][r];
    __syncthreads();
  }
}

// ---------------- LayerNorm over DV=128 per (b,s,h): wave-per-row ----------------
__global__ __launch_bounds__(256) void layernorm_kernel(const float* __restrict__ Ov,
                                                        __bf16* __restrict__ Y) {
  const int lane = threadIdx.x & 31;
  const long row = (((long)blockIdx.x * blockDim.x) + threadIdx.x) >> 5;
  const float* p = Ov + row * 128;
  float v[4], sum = 0.f;
#pragma unroll
  for (int t = 0; t < 4; ++t) { v[t] = p[lane + (t << 5)]; sum += v[t]; }
#pragma unroll
  for (int m = 16; m >= 1; m >>= 1) sum += __shfl_xor(sum, m, 32);
  const float mu = sum * (1.0f / 128.0f);
  float var = 0.f;
#pragma unroll
  for (int t = 0; t < 4; ++t) { const float d = v[t] - mu; var += d * d; }
#pragma unroll
  for (int m = 16; m >= 1; m >>= 1) var += __shfl_xor(var, m, 32);
  const float inv = rsqrtf(var * (1.0f / 128.0f) + EPS_);
#pragma unroll
  for (int t = 0; t < 4; ++t)
    Y[row * 128 + lane + (t << 5)] = (__bf16)((v[t] - mu) * inv);
}

// ---------------- launch ----------------
extern "C" void kernel_launch(void* const* d_in, const int* in_sizes, int n_in,
                              void* d_out, int out_size, void* d_ws, size_t ws_size,
                              hipStream_t stream) {
  (void)in_sizes; (void)n_in; (void)out_size; (void)ws_size;
  const float* x     = (const float*)d_in[0];
  const float* W_in  = (const float*)d_in[1];
  const float* W_emb = (const float*)d_in[2];
  const float* b_emb = (const float*)d_in[3];
  const float* W_out = (const float*)d_in[4];
  float* out = (float*)d_out;

  const long M = (long)B_ * S_;                    // 16384 token rows
  __bf16* xb   = (__bf16*)d_ws;                    // M x 2048 (reused as y after GEMM1)
  __bf16* Winb = xb + M * 2048;                    // 6144 x 2048
  __bf16* Wemb = Winb + 6144L * 2048;              // 2048 x 2048
  __bf16* Woub = Wemb + 2048L * 2048;              // 2048 x 2048
  __bf16* qkv  = Woub + 2048L * 2048;              // M x 6144
  __bf16* qe   = qkv + M * 6144;                   // M x 2048
  __bf16* ke   = qe + M * 2048;                    // M x 2048
  float*  o    = (float*)(ke + M * 2048);          // M x 2048 f32 (attn out, pre-LN)
  __bf16* y    = xb;                               // M x 2048 (xb dead after GEMM1)

  const dim3 blk(256);
  f32_to_bf16_kernel<<<dim3(32768), blk, 0, stream>>>(x, xb);
  f32_to_bf16_kernel<<<dim3(12288), blk, 0, stream>>>(W_in, Winb);
  f32_to_bf16_kernel<<<dim3(4096), blk, 0, stream>>>(W_emb, Wemb);
  f32_to_bf16_kernel<<<dim3(4096), blk, 0, stream>>>(W_out, Woub);

  // qkv = x @ W_in^T
  wmma_gemm<__bf16><<<dim3(48, 128), blk, 0, stream>>>(xb, 2048, Winb, nullptr,
                                                       qkv, 6144, 2048);
  // q/k embed: q|k slice of qkv @ W_embed^T + b
  wmma_gemm<__bf16><<<dim3(16, 128), blk, 0, stream>>>(qkv, 6144, Wemb, b_emb,
                                                       qe, 2048, 2048);
  wmma_gemm<__bf16><<<dim3(16, 128), blk, 0, stream>>>(qkv + 2048, 6144, Wemb, b_emb,
                                                       ke, 2048, 2048);
  // chunked linear attention with per-head decay (v = third slice of qkv)
  lightning_attn<<<dim3(B_ * H_), blk, 0, stream>>>(qe, ke, qkv + 4096, o);
  // per-head LayerNorm (262144 rows, 8 waves/block)
  layernorm_kernel<<<dim3(32768), blk, 0, stream>>>(o, y);
  // final projection: out = y @ W_out^T (f32 output)
  wmma_gemm<float><<<dim3(16, 128), blk, 0, stream>>>(y, 2048, Woub, nullptr,
                                                      out, 2048, 2048);
}
